// Block_13537736917617
// MI455X (gfx1250) — matrix operand order repaired and checked by reading, not compiled, not run on hardware
//
#include <hip/hip_runtime.h>
#include <hip/hip_bf16.h>
#include <math.h>

// ---------------- types ----------------
typedef __bf16 bf16_t;
typedef bf16_t v16bf __attribute__((ext_vector_type(16)));
typedef float  v8f   __attribute__((ext_vector_type(8)));

__device__ __forceinline__ unsigned short f2bf(float f) {
  unsigned u = __float_as_uint(f);
  unsigned r = 0x7FFFu + ((u >> 16) & 1u);   // round-to-nearest-even
  return (unsigned short)((u + r) >> 16);
}
__device__ __forceinline__ float bf2f(unsigned short h) {
  return __uint_as_float(((unsigned)h) << 16);
}
__device__ __forceinline__ unsigned pack2bf(float lo, float hi) {
  return (unsigned)f2bf(lo) | ((unsigned)f2bf(hi) << 16);
}
// lower 32 bits of a flat LDS address == LDS byte offset (aperture is 4GB aligned)
__device__ __forceinline__ unsigned lds_off(const void* p) {
  return (unsigned)(size_t)p;
}
// async 16-byte global->LDS copy (GV mode), tracked by ASYNCcnt
__device__ __forceinline__ void async_copy_b128(unsigned lds, const void* gptr) {
  asm volatile("global_load_async_to_lds_b128 %0, %1, off"
               :: "v"(lds), "v"(gptr) : "memory");
}
__device__ __forceinline__ void wait_async0() {
  asm volatile("s_wait_asynccnt 0" ::: "memory");
}

// A-fragment (16x32 bf16), lane m = lane&15, half = lane>>4.
// VGPR r holds K = 16*(r>>2) + 2*(r&3) + {0,1} + 8*half  (ISA 7.12.2)
__device__ __forceinline__ v16bf frag_a(const unsigned short* rowp, int half) {
  union { v16bf v; unsigned u[8]; } f;
#pragma unroll
  for (int r = 0; r < 8; ++r) {
    int k = ((r >> 2) << 4) + ((r & 3) << 1) + (half << 3);
    f.u[r] = *(const unsigned*)(rowp + k);
  }
  return f.v;
}
// B-fragment (32x16 bf16) read from a K-contiguous (transposed) LDS row for
// output column n = lane&15: VGPR r holds K = 16*(r>>2) + 2*(r&3) + {0,1} + 8*half
__device__ __forceinline__ v16bf frag_b(const unsigned short* colp, int half) {
  union { v16bf v; unsigned u[8]; } f;
#pragma unroll
  for (int r = 0; r < 8; ++r) {
    int k = ((r >> 2) << 4) + ((r & 3) << 1) + (half << 3);
    f.u[r] = *(const unsigned*)(colp + k);
  }
  return f.v;
}

// ---------------- RMSNorm: f32 -> bf16 ----------------
__global__ __launch_bounds__(256) void rmsnorm_kernel(
    const float* __restrict__ x, const float* __restrict__ w,
    unsigned short* __restrict__ out, int D) {
  __shared__ float red[256];
  const int row = blockIdx.x;
  const float* xr = x + (size_t)row * D;
  float ss = 0.f;
  for (int i = threadIdx.x; i < D; i += 256) { float v = xr[i]; ss += v * v; }
  red[threadIdx.x] = ss;
  __syncthreads();
  for (int s = 128; s > 0; s >>= 1) {
    if (threadIdx.x < s) red[threadIdx.x] += red[threadIdx.x + s];
    __syncthreads();
  }
  const float inv = 1.0f / (sqrtf(red[0] / (float)D) + 1e-5f);
  for (int i = threadIdx.x; i < D; i += 256)
    out[(size_t)row * D + i] = f2bf(w[i] * xr[i] * inv);
}

// ---------------- GEMM: C = A(bf16) x B(f32->bf16), f32 accumulate ----------------
// 128x128x32 block tile, double-buffered LDS; A staged by async-to-LDS DMA,
// B converted f32->bf16 on the VALU into a transposed LDS panel.
enum { EPI_BF16 = 0, EPI_RESID_F32 = 1, EPI_SWIGLU_BF16 = 2 };

template <int EPI>
__global__ __launch_bounds__(256) void gemm_bf16_kernel(
    const unsigned short* __restrict__ A, const float* __restrict__ B,
    void* __restrict__ Out, const float* __restrict__ Resid,
    const unsigned short* __restrict__ Gate, int M, int N, int K) {
  constexpr int BM = 128, BN = 128, BK = 32, LDA = 40, LDB = 40;
  __shared__ unsigned short As[2][BM * LDA];   // [m][k], row stride 80B (16B aligned)
  __shared__ unsigned short Bs[2][BN * LDB];   // transposed: [n][k]

  const int tid  = threadIdx.x;
  const int wave = tid >> 5, lane = tid & 31;
  const int half = lane >> 4, ln16 = lane & 15;
  const int waveM = wave & 1, waveN = wave >> 1;  // 2 x 4 wave grid
  const int m0 = blockIdx.y * BM, n0 = blockIdx.x * BN;

  // stage A tile (pure bf16 copy) via async DMA: 512 x 16B chunks
  auto stageA = [&](int buf, int kk0) {
#pragma unroll
    for (int i = tid; i < (BM * BK) / 8; i += 256) {
      int m = i >> 2, c = (i & 3) * 8;
      async_copy_b128(lds_off(&As[buf][m * LDA + c]),
                      &A[(size_t)(m0 + m) * K + kk0 + c]);
    }
  };
  // stage B tile with f32->bf16 conversion, transposed, packed k-pairs
  auto stageB = [&](int buf, int kk0) {
    for (int i = tid; i < (BK / 2) * BN; i += 256) {
      int n = i & 127, kp = i >> 7;  // kp in 0..15
      float f0 = B[(size_t)(kk0 + 2 * kp) * N + n0 + n];
      float f1 = B[(size_t)(kk0 + 2 * kp + 1) * N + n0 + n];
      *(unsigned*)&Bs[buf][n * LDB + 2 * kp] = pack2bf(f0, f1);
    }
  };

  v8f acc[4][2];
#pragma unroll
  for (int i = 0; i < 4; ++i)
#pragma unroll
    for (int j = 0; j < 2; ++j)
#pragma unroll
      for (int e = 0; e < 8; ++e) acc[i][j][e] = 0.f;

  stageA(0, 0);
  stageB(0, 0);
  wait_async0();
  __syncthreads();

  const int nk = K / BK;
  for (int step = 0; step < nk; ++step) {
    const int buf = step & 1;
    if (step + 1 < nk) {        // stage next tile while computing this one
      stageA(buf ^ 1, (step + 1) * BK);
      stageB(buf ^ 1, (step + 1) * BK);
    }
    const unsigned short* Ab = &As[buf][0];
    const unsigned short* Bb = &Bs[buf][0];

    v16bf bfrag[2];
#pragma unroll
    for (int ni = 0; ni < 2; ++ni)
      bfrag[ni] = frag_b(&Bb[(waveN * 32 + ni * 16 + ln16) * LDB], half);
#pragma unroll
    for (int mi = 0; mi < 4; ++mi) {
      v16bf afrag = frag_a(&Ab[(waveM * 64 + mi * 16 + ln16) * LDA], half);
#pragma unroll
      for (int ni = 0; ni < 2; ++ni)
        acc[mi][ni] = __builtin_amdgcn_wmma_f32_16x16x32_bf16(
            false, afrag, false, bfrag[ni], (short)0, acc[mi][ni], false, false);
    }
    if (step + 1 < nk) {
      wait_async0();            // next A tile landed
      __syncthreads();          // + everyone done reading current buffers
    }
  }

#pragma unroll
  for (int mi = 0; mi < 4; ++mi)
#pragma unroll
    for (int ni = 0; ni < 2; ++ni) {
      const int rbase = m0 + waveM * 64 + mi * 16 + half * 8;  // C row = r + 8*half
      const int col   = n0 + waveN * 32 + ni * 16 + ln16;
#pragma unroll
      for (int r = 0; r < 8; ++r) {
        const size_t idx = (size_t)(rbase + r) * N + col;
        const float vv = acc[mi][ni][r];
        if constexpr (EPI == EPI_BF16) {
          ((unsigned short*)Out)[idx] = f2bf(vv);
        } else if constexpr (EPI == EPI_RESID_F32) {
          ((float*)Out)[idx] = vv + Resid[idx];
        } else {  // fused SwiGLU: out = gate * sigmoid(acc)
          const float g  = bf2f(Gate[idx]);
          const float sg = 1.f / (1.f + __expf(-vv));
          ((unsigned short*)Out)[idx] = f2bf(g * sg);
        }
      }
    }
}

// ---------------- Flash attention (causal, GQA), bf16 WMMA ----------------
// grid: b(2) * h(16) * qtile(32); block: 128 threads = 4 waves, 16 q-rows/wave
__global__ __launch_bounds__(128) void attn_kernel(
    const unsigned short* __restrict__ q, const unsigned short* __restrict__ k,
    const unsigned short* __restrict__ v, unsigned short* __restrict__ o) {
  constexpr int QLD = 136, KLD = 136, VLD = 68, PLD = 68;  // 16B-aligned rows for async
  __shared__ unsigned short Qs[64 * QLD];
  __shared__ unsigned short Ks[64 * KLD];
  __shared__ unsigned short Vs[128 * VLD];       // transposed: [d][k]
  __shared__ unsigned short Ps[4 * 16 * PLD];    // wave-private P tiles

  const int tid = threadIdx.x;
  const int wave = tid >> 5, lane = tid & 31;
  const int half = lane >> 4, ln16 = lane & 15;

  const int qt  = blockIdx.x & 31;
  const int h   = (blockIdx.x >> 5) & 15;
  const int b   = blockIdx.x >> 9;
  const int kvh = h >> 2;                        // repeat_interleave: kv head = h/4
  const int q0  = qt * 64;

  // stage Q tile via async DMA: 1024 x 16B chunks
#pragma unroll
  for (int i = tid; i < 1024; i += 128) {
    int r = i >> 4, c = (i & 15) * 8;
    async_copy_b128(lds_off(&Qs[r * QLD + c]),
                    &q[(size_t)(b * 2048 + q0 + r) * 2048 + h * 128 + c]);
  }

  float m_r[8], l_r[8];
  v8f o_acc[8];
#pragma unroll
  for (int r = 0; r < 8; ++r) { m_r[r] = -3.0e38f; l_r[r] = 0.f; }
#pragma unroll
  for (int dt = 0; dt < 8; ++dt)
#pragma unroll
    for (int e = 0; e < 8; ++e) o_acc[dt][e] = 0.f;

  const float scale = 0.08838834764831845f;      // 1/sqrt(128)

  for (int kt = 0; kt <= qt; ++kt) {
    __syncthreads();                             // previous-iter readers done
    const int kr0 = kt * 64;
    // K tile: pure copy -> async DMA
#pragma unroll
    for (int i = tid; i < 1024; i += 128) {
      int r = i >> 4, c = (i & 15) * 8;
      async_copy_b128(lds_off(&Ks[r * KLD + c]),
                      &k[(size_t)(b * 2048 + kr0 + r) * 512 + kvh * 128 + c]);
    }
    // V tile: transpose -> 128-bit global reads, b16 LDS scatter
    for (int i = tid; i < 1024; i += 128) {
      int r = i >> 4, c0 = (i & 15) * 8;
      union { uint4 q4; unsigned short s[8]; } t;
      t.q4 = *(const uint4*)&v[(size_t)(b * 2048 + kr0 + r) * 512 + kvh * 128 + c0];
#pragma unroll
      for (int j = 0; j < 8; ++j) Vs[(c0 + j) * VLD + r] = t.s[j];
    }
    wait_async0();
    __syncthreads();

    // S = (Q K^T) for this wave's 16 q-rows: 4 tiles of 16x16, k-dim = DH in 4 steps
    v8f s[4];
#pragma unroll
    for (int ks = 0; ks < 4; ++ks) {
#pragma unroll
      for (int e = 0; e < 8; ++e) s[ks][e] = 0.f;
#pragma unroll
      for (int ds = 0; ds < 4; ++ds) {
        v16bf a  = frag_a(&Qs[(wave * 16 + ln16) * QLD + ds * 32], half);
        v16bf bb = frag_b(&Ks[(ks * 16 + ln16) * KLD + ds * 32], half);
        s[ks] = __builtin_amdgcn_wmma_f32_16x16x32_bf16(
            false, a, false, bb, (short)0, s[ks], false, false);
      }
    }

    // scale + causal mask (-1e9 additive, matching reference) + row max
    float mloc[8];
#pragma unroll
    for (int r = 0; r < 8; ++r) mloc[r] = -3.0e38f;
#pragma unroll
    for (int ks = 0; ks < 4; ++ks)
#pragma unroll
      for (int r = 0; r < 8; ++r) {
        float sv = s[ks][r] * scale;
        int qrow = q0 + wave * 16 + r + half * 8;
        int kcol = kr0 + ks * 16 + ln16;
        if (kcol > qrow) sv += -1e9f;
        s[ks][r] = sv;
        mloc[r] = fmaxf(mloc[r], sv);
      }
#pragma unroll
    for (int r = 0; r < 8; ++r)
      for (int msk = 8; msk >= 1; msk >>= 1)
        mloc[r] = fmaxf(mloc[r], __shfl_xor(mloc[r], msk, 16));

    float alpha[8];
#pragma unroll
    for (int r = 0; r < 8; ++r) {
      float mn = fmaxf(m_r[r], mloc[r]);
      alpha[r] = __expf(m_r[r] - mn);
      m_r[r] = mn;
      l_r[r] *= alpha[r];
    }
#pragma unroll
    for (int dt = 0; dt < 8; ++dt)
#pragma unroll
      for (int r = 0; r < 8; ++r) o_acc[dt][r] *= alpha[r];

    // P = exp(S - m); row sums; stage P (C-layout -> LDS -> A-layout)
    float rs[8];
#pragma unroll
    for (int r = 0; r < 8; ++r) rs[r] = 0.f;
#pragma unroll
    for (int ks = 0; ks < 4; ++ks)
#pragma unroll
      for (int r = 0; r < 8; ++r) {
        float p = __expf(s[ks][r] - m_r[r]);
        rs[r] += p;
        Ps[(wave * 16 + r + half * 8) * PLD + ks * 16 + ln16] = f2bf(p);
      }
#pragma unroll
    for (int r = 0; r < 8; ++r) {
      for (int msk = 8; msk >= 1; msk >>= 1) rs[r] += __shfl_xor(rs[r], msk, 16);
      l_r[r] += rs[r];
    }

    asm volatile("s_wait_dscnt 0" ::: "memory");  // wave-private LDS RAW

    // O += P @ V : 8 d-tiles, k-dim 64 in 2 WMMA steps
#pragma unroll
    for (int dt = 0; dt < 8; ++dt)
#pragma unroll
      for (int kk = 0; kk < 2; ++kk) {
        v16bf a  = frag_a(&Ps[(wave * 16 + ln16) * PLD + kk * 32], half);
        v16bf bb = frag_b(&Vs[(dt * 16 + ln16) * VLD + kk * 32], half);
        o_acc[dt] = __builtin_amdgcn_wmma_f32_16x16x32_bf16(
            false, a, false, bb, (short)0, o_acc[dt], false, false);
      }
  }

  // normalize and write O (bf16)
#pragma unroll
  for (int dt = 0; dt < 8; ++dt)
#pragma unroll
    for (int r = 0; r < 8; ++r) {
      float ov = o_acc[dt][r] / l_r[r];
      int row = b * 2048 + q0 + wave * 16 + r + half * 8;
      int col = h * 128 + dt * 16 + ln16;
      o[(size_t)row * 2048 + col] = f2bf(ov);
    }
}

// ---------------- host orchestration ----------------
extern "C" void kernel_launch(void* const* d_in, const int* in_sizes, int n_in,
                              void* d_out, int out_size, void* d_ws, size_t ws_size,
                              hipStream_t stream) {
  (void)in_sizes; (void)n_in; (void)out_size; (void)ws_size;
  const float* x   = (const float*)d_in[0];
  // d_in[1] attn_mask: identical to causal additive mask, applied analytically
  const float* n1w = (const float*)d_in[2];
  const float* n2w = (const float*)d_in[3];
  const float* Wq1 = (const float*)d_in[4];
  const float* Wq2 = (const float*)d_in[5];
  const float* Wk1 = (const float*)d_in[6];
  const float* Wk2 = (const float*)d_in[7];
  const float* Wv1 = (const float*)d_in[8];
  const float* Wv2 = (const float*)d_in[9];
  const float* Wo  = (const float*)d_in[10];
  const float* WuA = (const float*)d_in[11];
  const float* WuB = (const float*)d_in[12];
  const float* Wd  = (const float*)d_in[13];
  float* out = (float*)d_out;

  const int BT = 2 * 2048;  // 4096 token rows
  char* ws = (char*)d_ws;
  size_t off = 0;
  auto take = [&](size_t bytes) -> void* {
    void* p = ws + off;
    off += (bytes + 255) & ~(size_t)255;
    return p;
  };
  unsigned short* h1   = (unsigned short*)take((size_t)BT * 2048 * 2);
  unsigned short* qlat = (unsigned short*)take((size_t)BT * 512 * 2);
  unsigned short* qb   = (unsigned short*)take((size_t)BT * 2048 * 2);
  unsigned short* klat = (unsigned short*)take((size_t)BT * 256 * 2);
  unsigned short* kb   = (unsigned short*)take((size_t)BT * 512 * 2);
  unsigned short* vlat = (unsigned short*)take((size_t)BT * 256 * 2);
  unsigned short* vb   = (unsigned short*)take((size_t)BT * 512 * 2);
  unsigned short* ao   = (unsigned short*)take((size_t)BT * 2048 * 2);
  float*          x1   = (float*)take((size_t)BT * 2048 * 4);
  unsigned short* h2   = (unsigned short*)take((size_t)BT * 2048 * 2);
  unsigned short* gbuf = (unsigned short*)take((size_t)BT * 8192 * 2);

  // h1 = rmsnorm(x) in bf16
  rmsnorm_kernel<<<BT, 256, 0, stream>>>(x, n1w, h1, 2048);

  // Low-rank Q/K/V
  gemm_bf16_kernel<EPI_BF16><<<dim3(4, 32), 256, 0, stream>>>(h1, Wq1, qlat, nullptr, nullptr, BT, 512, 2048);
  gemm_bf16_kernel<EPI_BF16><<<dim3(16, 32), 256, 0, stream>>>(qlat, Wq2, qb, nullptr, nullptr, BT, 2048, 512);
  gemm_bf16_kernel<EPI_BF16><<<dim3(2, 32), 256, 0, stream>>>(h1, Wk1, klat, nullptr, nullptr, BT, 256, 2048);
  gemm_bf16_kernel<EPI_BF16><<<dim3(4, 32), 256, 0, stream>>>(klat, Wk2, kb, nullptr, nullptr, BT, 512, 256);
  gemm_bf16_kernel<EPI_BF16><<<dim3(2, 32), 256, 0, stream>>>(h1, Wv1, vlat, nullptr, nullptr, BT, 256, 2048);
  gemm_bf16_kernel<EPI_BF16><<<dim3(4, 32), 256, 0, stream>>>(vlat, Wv2, vb, nullptr, nullptr, BT, 512, 256);

  // Causal GQA attention
  attn_kernel<<<2 * 16 * 32, 128, 0, stream>>>(qb, kb, vb, ao);

  // x1 = x + O @ Wo  (f32 residual stream)
  gemm_bf16_kernel<EPI_RESID_F32><<<dim3(16, 32), 256, 0, stream>>>(ao, Wo, x1, x, nullptr, BT, 2048, 2048);

  // SwiGLU MLP
  rmsnorm_kernel<<<BT, 256, 0, stream>>>(x1, n2w, h2, 2048);
  gemm_bf16_kernel<EPI_BF16><<<dim3(64, 32), 256, 0, stream>>>(h2, WuA, gbuf, nullptr, nullptr, BT, 8192, 2048);
  gemm_bf16_kernel<EPI_SWIGLU_BF16><<<dim3(64, 32), 256, 0, stream>>>(h2, WuB, gbuf, nullptr, gbuf, BT, 8192, 2048);

  // out = x1 + g @ W_down
  gemm_bf16_kernel<EPI_RESID_F32><<<dim3(16, 32), 256, 0, stream>>>(gbuf, Wd, out, x1, nullptr, BT, 2048, 8192);
}
